// SelfAttentionEinops_76974403879345
// MI455X (gfx1250) — matrix operand order repaired and checked by reading, not compiled, not run on hardware
//
#include <hip/hip_runtime.h>

// ---------------------------------------------------------------------------
// CDNA5 (gfx1250) attention block:  Q/K/V proj -> RoPE(K,V) -> GQA causal
// attention -> out proj.  All matmuls via V_WMMA_F32_16X16X32_BF16 (wave32).
// GEMM: 128x128 block tile, 8 waves (32x64 each), double-buffered LDS.
// ---------------------------------------------------------------------------

typedef __bf16 bf16;
typedef __attribute__((ext_vector_type(16))) __bf16 v16bf;
typedef __attribute__((ext_vector_type(8)))  __bf16 v8bf;
typedef __attribute__((ext_vector_type(8)))  float  v8f;

union FragBF { v16bf v; v8bf h[2]; };

__device__ __forceinline__ v8f wmma_bf16(v16bf a, v16bf b, v8f c) {
  // D = A(16x32 bf16) * B(32x16 bf16) + C(16x16 f32)
  return __builtin_amdgcn_wmma_f32_16x16x32_bf16(false, a, false, b, (short)0, c,
                                                 false, false);
}

__device__ __forceinline__ v8f vzero8() {
  v8f z;
#pragma unroll
  for (int i = 0; i < 8; ++i) z[i] = 0.0f;
  return z;
}

// ------------------------------ prep kernels -------------------------------

__global__ void k_cvt_bf16(const float* __restrict__ in, bf16* __restrict__ out,
                           int n) {
  int i = blockIdx.x * 256 + threadIdx.x;
  if (i < n) out[i] = (bf16)in[i];
}

// W (K x N, row-major f32)  ->  Wt (N x K, row-major bf16)
__global__ void k_tr_bf16(const float* __restrict__ W, bf16* __restrict__ Wt,
                          int K, int N) {
  int i = blockIdx.x * 256 + threadIdx.x;
  if (i >= K * N) return;
  int k = i / N, n = i % N;
  Wt[(size_t)n * K + k] = (bf16)W[i];
}

// ------------------------------- GEMM (bf16) -------------------------------
// C(MxN) = A(MxK, bf16 row-major) * Bt(NxK, bf16 row-major)^T
// Block tile 128x128, 8 waves in 4x2; each wave computes 32x64 (2x4 WMMA).
// Double-buffered LDS staging so global loads overlap the WMMA block.
__launch_bounds__(256)
__global__ void k_gemm(const bf16* __restrict__ A, const bf16* __restrict__ Bt,
                       float* __restrict__ Cf, bf16* __restrict__ Cb,
                       int M, int N, int K) {
  __shared__ __align__(16) bf16 At[2][128 * 32];
  __shared__ __align__(16) bf16 Bs[2][128 * 32];
  const int tid  = threadIdx.x;
  const int lane = tid & 31, wave = tid >> 5;
  const int ln = lane & 15, hf = lane >> 4;
  const int wm = wave >> 1, wn = wave & 1;          // 4x2 wave grid
  const int m0 = blockIdx.y * 128, n0 = blockIdx.x * 128;

  v8f acc[2][4];
#pragma unroll
  for (int i = 0; i < 2; ++i)
#pragma unroll
    for (int j = 0; j < 4; ++j) acc[i][j] = vzero8();

  v8bf ra[2], rb[2];                                 // staged global data

  auto gload = [&](int k0) {
#pragma unroll
    for (int i = 0; i < 2; ++i) {
      int off = (tid * 2 + i) * 8;                   // 128x32 tile element
      int r = off >> 5, c = off & 31;
      ra[i] = *(const v8bf*)(A + (size_t)(m0 + r) * K + k0 + c);
      rb[i] = *(const v8bf*)(Bt + (size_t)(n0 + r) * K + k0 + c);
    }
  };
  auto sstore = [&](int buf) {
#pragma unroll
    for (int i = 0; i < 2; ++i) {
      int off = (tid * 2 + i) * 8;
      *(v8bf*)(&At[buf][off]) = ra[i];
      *(v8bf*)(&Bs[buf][off]) = rb[i];
    }
  };
  auto compute = [&](int buf) {
    FragBF fa[2], fb[4];
#pragma unroll
    for (int i = 0; i < 2; ++i) {
      // A fragment (ISA 16-bit A layout): lane m=ln, VGPR0-3 -> K[8h..8h+8),
      // VGPR4-7 -> K[16+8h..24+8h)
      const bf16* ar = &At[buf][(wm * 32 + i * 16 + ln) * 32];
      fa[i].h[0] = *(const v8bf*)(ar + 8 * hf);
      fa[i].h[1] = *(const v8bf*)(ar + 16 + 8 * hf);
    }
#pragma unroll
    for (int j = 0; j < 4; ++j) {
      // B fragment: lane n=ln, VGPR0-3 -> K[8h..8h+8), VGPR4-7 -> K[16+8h..24+8h)
      const bf16* br = &Bs[buf][(wn * 64 + j * 16 + ln) * 32];
      fb[j].h[0] = *(const v8bf*)(br + 8 * hf);
      fb[j].h[1] = *(const v8bf*)(br + 16 + 8 * hf);
    }
#pragma unroll
    for (int i = 0; i < 2; ++i)
#pragma unroll
      for (int j = 0; j < 4; ++j)
        acc[i][j] = wmma_bf16(fa[i].v, fb[j].v, acc[i][j]);
  };

  gload(0);
  sstore(0);
  __syncthreads();
  int cur = 0;
  for (int k0 = 32; k0 < K; k0 += 32) {
    gload(k0);                      // next tile -> regs (overlaps WMMAs)
    {                               // L2 prefetch for the tile after next
      int kp = k0 + 32;
      if (kp < K) {
        int off = (tid * 2) * 8;
        int r = off >> 5, c = off & 31;
        __builtin_prefetch(A + (size_t)(m0 + r) * K + kp + c, 0, 0);
        __builtin_prefetch(Bt + (size_t)(n0 + r) * K + kp + c, 0, 0);
      }
    }
    compute(cur);                   // WMMAs on current LDS buffer
    sstore(cur ^ 1);                // stage next tile into the other buffer
    __syncthreads();
    cur ^= 1;
  }
  compute(cur);

  // C layout: VGPR r -> row (r + 8*hf), col = ln
#pragma unroll
  for (int i = 0; i < 2; ++i)
#pragma unroll
    for (int j = 0; j < 4; ++j)
#pragma unroll
      for (int r = 0; r < 8; ++r) {
        int row = m0 + wm * 32 + i * 16 + r + 8 * hf;
        int col = n0 + wn * 64 + j * 16 + ln;
        if (Cb) Cb[(size_t)row * N + col] = (bf16)acc[i][j][r];
        else    Cf[(size_t)row * N + col] = acc[i][j][r];
      }
}

// ------------------------------ RoPE on K, V -------------------------------
// kv: (B*S) x 1024 fp32  (cols 0..511 = K heads, 512..1023 = V heads)
// Kb: [b][s][kvh*128+d] bf16 ; Vt: [b][kvh][d][s] bf16 (transposed for P*V)
__global__ void k_rope(const float* __restrict__ kv,
                       const float* __restrict__ cosT,
                       const float* __restrict__ sinT,
                       bf16* __restrict__ Kb, bf16* __restrict__ Vt) {
  int t = blockIdx.x * 256 + threadIdx.x;          // B*S*4*64 = 524288
  int j   = t & 63;
  int kvh = (t >> 6) & 3;
  int s   = (t >> 8) & 1023;
  int b   = t >> 18;
  size_t base = ((size_t)(b * 1024 + s)) * 1024 + kvh * 128 + 2 * j;
  float c  = cosT[s * 64 + j];
  float sn = sinT[s * 64 + j];
  float k0 = kv[base],       k1 = kv[base + 1];
  float v0 = kv[base + 512], v1 = kv[base + 513];
  size_t ko = ((size_t)(b * 1024 + s)) * 512 + kvh * 128 + 2 * j;
  Kb[ko]     = (bf16)(k0 * c - k1 * sn);
  Kb[ko + 1] = (bf16)(k0 * sn + k1 * c);
  size_t vo = ((size_t)((b * 4 + kvh) * 128 + 2 * j)) * 1024 + s;
  Vt[vo]        = (bf16)(v0 * c - v1 * sn);
  Vt[vo + 1024] = (bf16)(v0 * sn + v1 * c);
}

// ------------------------- Flash attention (causal) ------------------------
// grid(S/128, N_HEADS, B); 8 waves/block, 16 queries per wave.
__launch_bounds__(256)
__global__ void k_attn(const bf16* __restrict__ Q, const bf16* __restrict__ Kb,
                       const bf16* __restrict__ Vt, bf16* __restrict__ Y) {
  __shared__ __align__(16) bf16 pbuf[8 * 16 * 32];   // per-wave P staging
  const int lane = threadIdx.x & 31, wave = threadIdx.x >> 5;
  const int ln = lane & 15, hf = lane >> 4;
  const int b = blockIdx.z, h = blockIdx.y, kvh = h >> 2;
  const int qbase = blockIdx.x * 128 + wave * 16;
  const float scale = 0.088388347648318447f;         // 1/sqrt(128)

  // Q A-fragments for the 4 d-chunks of 32
  FragBF qf[4];
  const bf16* qrow = Q + ((size_t)(b * 1024 + qbase + ln)) * 2048 + h * 128;
#pragma unroll
  for (int c = 0; c < 4; ++c) {
    qf[c].h[0] = *(const v8bf*)(qrow + c * 32 + 8 * hf);
    qf[c].h[1] = *(const v8bf*)(qrow + c * 32 + 16 + 8 * hf);
  }

  v8f acc[8];
  float mrun[8], lrun[8];
#pragma unroll
  for (int dt = 0; dt < 8; ++dt) acc[dt] = vzero8();
#pragma unroll
  for (int r = 0; r < 8; ++r) { mrun[r] = -1e30f; lrun[r] = 0.0f; }

  bf16* pb = pbuf + wave * 512;
  const bf16* vbase = Vt + ((size_t)(b * 4 + kvh)) * 128 * 1024;
  const int kend = qbase + 15;                       // causal early exit

  for (int ks = 0; ks <= kend; ks += 32) {
    // ---- scores S = Q * K^T for 2 key sub-tiles of 16 ----
    v8f sfr[2];
#pragma unroll
    for (int t = 0; t < 2; ++t) {
      v8f s = vzero8();
      const bf16* kp =
          Kb + ((size_t)(b * 1024 + ks + t * 16 + ln)) * 512 + kvh * 128;
#pragma unroll
      for (int c = 0; c < 4; ++c) {
        FragBF kf;
        kf.h[0] = *(const v8bf*)(kp + c * 32 + 8 * hf);
        kf.h[1] = *(const v8bf*)(kp + c * 32 + 16 + 8 * hf);
        s = wmma_bf16(qf[c].v, kf.v, s);
      }
      sfr[t] = s;
    }
    // ---- online softmax (row stats across 16 lanes via shfl_xor) ----
#pragma unroll
    for (int r = 0; r < 8; ++r) {
      int rowq = qbase + r + 8 * hf;
      float mx = -1e30f;
#pragma unroll
      for (int t = 0; t < 2; ++t) {
        int colk = ks + t * 16 + ln;
        float v = sfr[t][r] * scale;
        if (colk > rowq) v = -1e30f;                 // causal mask
        sfr[t][r] = v;
        mx = fmaxf(mx, v);
      }
      mx = fmaxf(mx, __shfl_xor(mx, 1));
      mx = fmaxf(mx, __shfl_xor(mx, 2));
      mx = fmaxf(mx, __shfl_xor(mx, 4));
      mx = fmaxf(mx, __shfl_xor(mx, 8));
      float mnew  = fmaxf(mrun[r], mx);
      float alpha = __expf(mrun[r] - mnew);
      mrun[r] = mnew;
      float sum = 0.0f;
#pragma unroll
      for (int t = 0; t < 2; ++t) {
        float p = __expf(sfr[t][r] - mnew);
        sfr[t][r] = p;
        sum += p;
      }
      sum += __shfl_xor(sum, 1);
      sum += __shfl_xor(sum, 2);
      sum += __shfl_xor(sum, 4);
      sum += __shfl_xor(sum, 8);
      lrun[r] = lrun[r] * alpha + sum;
#pragma unroll
      for (int dt = 0; dt < 8; ++dt) acc[dt][r] *= alpha;
    }
    // ---- P: C-layout -> A-layout via per-wave LDS (same-wave DS in-order) --
#pragma unroll
    for (int r = 0; r < 8; ++r) {
      int row = r + 8 * hf;
      pb[row * 32 + ln]      = (bf16)sfr[0][r];
      pb[row * 32 + 16 + ln] = (bf16)sfr[1][r];
    }
    FragBF pf;
    pf.h[0] = *(const v8bf*)(pb + ln * 32 + 8 * hf);
    pf.h[1] = *(const v8bf*)(pb + ln * 32 + 16 + 8 * hf);
    // ---- O += P * V  (Vt rows contiguous along keys) ----
#pragma unroll
    for (int dt = 0; dt < 8; ++dt) {
      const bf16* vp = vbase + (size_t)(dt * 16 + ln) * 1024 + ks + 8 * hf;
      FragBF vf;
      vf.h[0] = *(const v8bf*)(vp);
      vf.h[1] = *(const v8bf*)(vp + 16);
      acc[dt] = wmma_bf16(pf.v, vf.v, acc[dt]);
    }
  }

  // y[b][s][h*128+d] bf16, normalized by row sums
#pragma unroll
  for (int dt = 0; dt < 8; ++dt)
#pragma unroll
    for (int r = 0; r < 8; ++r) {
      size_t row = (size_t)(b * 1024 + qbase + r + 8 * hf);
      Y[row * 2048 + h * 128 + dt * 16 + ln] = (bf16)(acc[dt][r] / lrun[r]);
    }
}

// --------------------------------- launcher --------------------------------

extern "C" void kernel_launch(void* const* d_in, const int* in_sizes, int n_in,
                              void* d_out, int out_size, void* d_ws,
                              size_t ws_size, hipStream_t stream) {
  const float* x    = (const float*)d_in[0];   // (2,1024,2048)
  const float* cosT = (const float*)d_in[1];   // (1024,64)
  const float* sinT = (const float*)d_in[2];   // (1024,64)
  const float* wq   = (const float*)d_in[3];   // (2048,2048)
  const float* wk   = (const float*)d_in[4];   // (2048,512)
  const float* wv   = (const float*)d_in[5];   // (2048,512)
  const float* wo   = (const float*)d_in[6];   // (2048,2048)
  float* out = (float*)d_out;

  const int M = 2048;       // B*S
  const int DIM = 2048, KVD = 512;

  char* ws = (char*)d_ws;
  size_t o = 0;
  bf16* xb   = (bf16*)(ws + o); o += (size_t)M * DIM * 2;        // 8 MB
  bf16* wqt  = (bf16*)(ws + o); o += (size_t)DIM * DIM * 2;      // 8 MB
  bf16* wkvt = (bf16*)(ws + o); o += (size_t)2 * KVD * DIM * 2;  // 4 MB
  bf16* wot  = (bf16*)(ws + o); o += (size_t)DIM * DIM * 2;      // 8 MB
  bf16* qb   = (bf16*)(ws + o); o += (size_t)M * DIM * 2;        // 8 MB
  float* kvf = (float*)(ws + o); o += (size_t)M * 1024 * 4;      // 8 MB
  bf16* kb   = (bf16*)(ws + o); o += (size_t)M * KVD * 2;        // 2 MB
  bf16* vtb  = (bf16*)(ws + o); o += (size_t)M * KVD * 2;        // 2 MB
  bf16* yb   = (bf16*)(ws + o); o += (size_t)M * DIM * 2;        // 8 MB

  // fp32 -> bf16 conversions / weight transposes
  k_cvt_bf16<<<(M * DIM + 255) / 256, 256, 0, stream>>>(x, xb, M * DIM);
  k_tr_bf16<<<(DIM * DIM + 255) / 256, 256, 0, stream>>>(wq, wqt, DIM, DIM);
  k_tr_bf16<<<(DIM * KVD + 255) / 256, 256, 0, stream>>>(wk, wkvt, DIM, KVD);
  k_tr_bf16<<<(DIM * KVD + 255) / 256, 256, 0, stream>>>(
      wv, wkvt + (size_t)KVD * DIM, DIM, KVD);
  k_tr_bf16<<<(DIM * DIM + 255) / 256, 256, 0, stream>>>(wo, wot, DIM, DIM);

  // Q = x @ wq (bf16 out) ; [K|V] = x @ [wk|wv] (fp32 out, pre-RoPE)
  k_gemm<<<dim3(DIM / 128, M / 128), 256, 0, stream>>>(
      xb, wqt, nullptr, qb, M, DIM, DIM);
  k_gemm<<<dim3(1024 / 128, M / 128), 256, 0, stream>>>(
      xb, wkvt, kvf, nullptr, M, 1024, DIM);

  // RoPE on K and V; V written transposed for the P*V WMMA
  k_rope<<<(2 * 1024 * 4 * 64) / 256, 256, 0, stream>>>(kvf, cosT, sinT, kb,
                                                        vtb);

  // causal GQA attention
  k_attn<<<dim3(1024 / 128, 16, 2), 256, 0, stream>>>(qb, kb, vtb, yb);

  // out = y @ wo (fp32 out)
  k_gemm<<<dim3(DIM / 128, M / 128), 256, 0, stream>>>(
      yb, wot, out, nullptr, M, DIM, DIM);
}
